// WindowAttention_75453985457437
// MI455X (gfx1250) — matrix operand order repaired and checked by reading, not compiled, not run on hardware
//
#include <hip/hip_runtime.h>

typedef __attribute__((ext_vector_type(16))) __bf16 v16bf;
typedef __attribute__((ext_vector_type(8)))  __bf16 v8bf;
typedef __attribute__((ext_vector_type(4)))  __bf16 v4bf;
typedef __attribute__((ext_vector_type(8)))  float  v8f;

#define LOG2E 1.4426950408889634f

// ---------------- prologue: expand relative-position bias -------------------
// bw[head][q][k] = bias_table[rpi(q,k)][head] * log2(e)   (1.57 MB, L2-resident)
__global__ void bias_expand_kernel(const float* __restrict__ bt,
                                   float* __restrict__ bw) {
  int i = blockIdx.x * 256 + threadIdx.x;
  if (i >= 6 * 256 * 256) return;
  int kk = i & 255;
  int qq = (i >> 8) & 255;
  int hd = i >> 16;
  int dy = (qq >> 4) - (kk >> 4) + 15;
  int dx = (qq & 15) - (kk & 15) + 15;
  bw[i] = bt[(dy * 31 + dx) * 6 + hd] * LOG2E;
}

// Load a WMMA 16-bit operand: two contiguous 16B LDS loads -> v16bf
__device__ __forceinline__ v16bf ld_op(const __bf16* lo, const __bf16* hi) {
  v8bf a = *(const v8bf*)lo;
  v8bf b = *(const v8bf*)hi;
  return __builtin_shufflevector(a, b, 0,1,2,3,4,5,6,7,8,9,10,11,12,13,14,15);
}

// 3-way max in one VOP3 op (no canonicalization)
__device__ __forceinline__ float vmax3(float a, float b, float c) {
  float d;
  asm("v_max3_num_f32 %0, %1, %2, %3" : "=v"(d) : "v"(a), "v"(b), "v"(c));
  return d;
}

// ---- DPP16 all-reduce within each 16-lane half-group, one VALU op/stage ----
// butterfly: xor1, xor2, (xor4 via half-mirror), (xor8 via mirror); DPP never
// crosses the 16-lane boundary, so the two half-groups stay independent.
__device__ __forceinline__ float rowmax16(float v) {
  float d;
  asm("v_max_num_f32_dpp %0, %1, %1 quad_perm:[1,0,3,2] row_mask:0xf bank_mask:0xf bound_ctrl:1"
      : "=v"(d) : "v"(v));
  asm("v_max_num_f32_dpp %0, %1, %1 quad_perm:[2,3,0,1] row_mask:0xf bank_mask:0xf bound_ctrl:1"
      : "=v"(v) : "v"(d));
  asm("v_max_num_f32_dpp %0, %1, %1 row_half_mirror row_mask:0xf bank_mask:0xf bound_ctrl:1"
      : "=v"(d) : "v"(v));
  asm("v_max_num_f32_dpp %0, %1, %1 row_mirror row_mask:0xf bank_mask:0xf bound_ctrl:1"
      : "=v"(v) : "v"(d));
  return v;
}
__device__ __forceinline__ float rowsum16(float v) {
  float d;
  asm("v_add_f32_dpp %0, %1, %1 quad_perm:[1,0,3,2] row_mask:0xf bank_mask:0xf bound_ctrl:1"
      : "=v"(d) : "v"(v));
  asm("v_add_f32_dpp %0, %1, %1 quad_perm:[2,3,0,1] row_mask:0xf bank_mask:0xf bound_ctrl:1"
      : "=v"(v) : "v"(d));
  asm("v_add_f32_dpp %0, %1, %1 row_half_mirror row_mask:0xf bank_mask:0xf bound_ctrl:1"
      : "=v"(d) : "v"(v));
  asm("v_add_f32_dpp %0, %1, %1 row_mirror row_mask:0xf bank_mask:0xf bound_ctrl:1"
      : "=v"(v) : "v"(d));
  return v;
}

// ------------- main kernel: one workgroup per (window, head) ----------------
__launch_bounds__(256)
__global__ void win_attn_kernel(const float* __restrict__ qg,
                                const float* __restrict__ kg,
                                const float* __restrict__ vg,
                                const float* __restrict__ bw,
                                float* __restrict__ outg) {
  extern __shared__ __align__(16) unsigned char smem_raw[];
  __bf16* sQ  = (__bf16*)smem_raw;   // 256 x 40 (stride pad for banks)
  __bf16* sK  = sQ + 256 * 40;       // 256 x 40
  __bf16* sVt = sK + 256 * 40;       // 32 x 264 (V transposed)
  __bf16* sP  = sVt + 32 * 264;      // 8 waves x 16 x 40 scratch

  const int wgid = blockIdx.x;       // 0..1535
  const int head = wgid % 6;
  const int win  = wgid / 6;
  const int wx = win & 7, wy = (win >> 3) & 7, bi = win >> 6;
  const int tid  = threadIdx.x;
  const int lane = tid & 31, wid = tid >> 5;
  const int lhi  = lane >> 4;        // 0/1: which half-group
  const int llo  = lane & 15;

  // ---- stage Q,K row-major and V transposed into LDS as bf16 ----
  {
    const int ty = tid >> 4, tx = tid & 15;
    const size_t goff =
        (((size_t)bi * 128 + (wy * 16 + ty)) * 128 + (wx * 16 + tx)) * 192 +
        (size_t)head * 32;
    const float4* qv = (const float4*)(qg + goff);
    const float4* kv = (const float4*)(kg + goff);
    const float4* vv = (const float4*)(vg + goff);
    const float qs = 0.1767766952966369f * LOG2E;  // 32^-0.5 * log2(e)
#pragma unroll
    for (int i = 0; i < 8; ++i) {
      float4 f = qv[i];
      v4bf b = { (__bf16)(f.x * qs), (__bf16)(f.y * qs),
                 (__bf16)(f.z * qs), (__bf16)(f.w * qs) };
      *(v4bf*)(sQ + tid * 40 + i * 4) = b;
    }
#pragma unroll
    for (int i = 0; i < 8; ++i) {
      float4 f = kv[i];
      v4bf b = { (__bf16)f.x, (__bf16)f.y, (__bf16)f.z, (__bf16)f.w };
      *(v4bf*)(sK + tid * 40 + i * 4) = b;
    }
#pragma unroll
    for (int i = 0; i < 8; ++i) {
      float4 f = vv[i];
      sVt[(4 * i + 0) * 264 + tid] = (__bf16)f.x;
      sVt[(4 * i + 1) * 264 + tid] = (__bf16)f.y;
      sVt[(4 * i + 2) * 264 + tid] = (__bf16)f.z;
      sVt[(4 * i + 3) * 264 + tid] = (__bf16)f.w;
    }
  }
  __syncthreads();

  const float* bh = bw + (size_t)head * 65536;
  __bf16* pw = sP + wid * (16 * 40);

  for (int qi = 0; qi < 2; ++qi) {
    const int q0 = wid * 32 + qi * 16;

    // A operand (Q rows q0..q0+15): lane<16 -> K 0..15; lane>=16 -> 16..31
    const __bf16* qrow = sQ + (q0 + llo) * 40 + lhi * 16;
    v16bf Aq = ld_op(qrow, qrow + 8);

    v8f O0 = {0,0,0,0,0,0,0,0};
    v8f O1 = {0,0,0,0,0,0,0,0};
    float mrun[8], lrun[8];
#pragma unroll
    for (int r = 0; r < 8; ++r) { mrun[r] = -3.0e38f; lrun[r] = 0.f; }

    const float* bbase = bh + (size_t)(q0 + 8 * lhi) * 256 + llo;

#pragma unroll 2
    for (int j = 0; j < 8; ++j) {
      // ---- bias tile straight into the WMMA C operand (D = A*B + C) ----
      const float* bp = bbase + 32 * j;
      v8f C0, C1;
#pragma unroll
      for (int r = 0; r < 8; ++r) {
        C0[r] = bp[r * 256];
        C1[r] = bp[r * 256 + 16];
      }

      // ---- S = Q * K^T + bias for key block [32j, 32j+32) ----
      const __bf16* kr0 = sK + (32 * j + llo) * 40 + lhi * 16;
      const __bf16* kr1 = sK + (32 * j + 16 + llo) * 40 + lhi * 16;
      v16bf Bk0 = ld_op(kr0, kr0 + 8);
      v16bf Bk1 = ld_op(kr1, kr1 + 8);
      v8f S0 = __builtin_amdgcn_wmma_f32_16x16x32_bf16(false, Aq, false, Bk0,
                                                       (short)0, C0, false, false);
      v8f S1 = __builtin_amdgcn_wmma_f32_16x16x32_bf16(false, Aq, false, Bk1,
                                                       (short)0, C1, false, false);

      // ---- online softmax (base-2 domain), fused DPP reductions ----
      // mrun is row-uniform, so it can be folded before the cross-lane max.
#pragma unroll
      for (int r = 0; r < 8; ++r) {
        float mn = rowmax16(vmax3(S0[r], S1[r], mrun[r]));
        float sc = __builtin_amdgcn_exp2f(mrun[r] - mn);
        mrun[r] = mn;
        float p0 = __builtin_amdgcn_exp2f(S0[r] - mn);
        float p1 = __builtin_amdgcn_exp2f(S1[r] - mn);
        S0[r] = p0; S1[r] = p1;
        lrun[r] = lrun[r] * sc + p0 + p1;   // per-lane partial row sum
        O0[r] *= sc;
        O1[r] *= sc;
      }

      // ---- C-layout P -> A-layout bf16 via per-wave LDS scratch ----
#pragma unroll
      for (int r = 0; r < 8; ++r) {
        int m = r + 8 * lhi;
        pw[m * 40 + llo]      = (__bf16)S0[r];
        pw[m * 40 + 16 + llo] = (__bf16)S1[r];
      }
      asm volatile("" ::: "memory");  // DS ops are in-order within a wave
      const __bf16* prow = pw + llo * 40 + lhi * 16;
      v16bf Ap = ld_op(prow, prow + 8);

      // ---- O += P * V_block  (V transposed: contiguous B operand loads) ----
      const __bf16* vr0 = sVt + llo * 264 + 32 * j + lhi * 16;
      const __bf16* vr1 = vr0 + 16 * 264;
      v16bf Bv0 = ld_op(vr0, vr0 + 8);
      v16bf Bv1 = ld_op(vr1, vr1 + 8);
      O0 = __builtin_amdgcn_wmma_f32_16x16x32_bf16(false, Ap, false, Bv0,
                                                   (short)0, O0, false, false);
      O1 = __builtin_amdgcn_wmma_f32_16x16x32_bf16(false, Ap, false, Bv1,
                                                   (short)0, O1, false, false);
    }

    // ---- epilogue: reduce row sums once, normalize, scatter to (b,h,w,c) ----
#pragma unroll
    for (int r = 0; r < 8; ++r) {
      float l = rowsum16(lrun[r]);
      float inv = __builtin_amdgcn_rcpf(l);
      int m = q0 + r + 8 * lhi;
      int ty = m >> 4, tx = m & 15;
      size_t ob = (((size_t)bi * 128 + (wy * 16 + ty)) * 128 + (wx * 16 + tx)) * 192 +
                  (size_t)head * 32;
      outg[ob + llo]      = O0[r] * inv;
      outg[ob + 16 + llo] = O1[r] * inv;
    }
  }
}

extern "C" void kernel_launch(void* const* d_in, const int* in_sizes, int n_in,
                              void* d_out, int out_size, void* d_ws, size_t ws_size,
                              hipStream_t stream) {
  (void)in_sizes; (void)n_in; (void)out_size; (void)ws_size;
  const float* q  = (const float*)d_in[0];
  const float* k  = (const float*)d_in[1];
  const float* v  = (const float*)d_in[2];
  const float* bt = (const float*)d_in[3];
  // d_in[4]=h, d_in[5]=w are fixed at 128 by the reference setup.
  float* out = (float*)d_out;
  float* bw  = (float*)d_ws;  // 6*256*256 f32 = 1.57 MB

  bias_expand_kernel<<<(6 * 256 * 256 + 255) / 256, 256, 0, stream>>>(bt, bw);

  const int nblocks = 256 * 6;            // windows * heads
  const size_t lds = (256*40 + 256*40 + 32*264 + 8*16*40) * sizeof(__bf16); // 68096 B
  win_attn_kernel<<<nblocks, 256, lds, stream>>>(q, k, v, bw, out);
}
